// HawkBlock_8572754723032
// MI455X (gfx1250) — matrix operand order repaired and checked by reading, not compiled, not run on hardware
//
#include <hip/hip_runtime.h>
#include <hip/hip_bf16.h>

// ---------------------------------------------------------------------------
// CDNA5 (gfx1250) Hawk block. bf16 WMMA GEMMs (f32 accum) + fp32 scan/norms.
// Data movement: async global->LDS copies (ASYNCcnt) for activations and a
// Tensor Data Mover descriptor load (TENSORcnt) for weight tiles, with
// fallbacks guarded by __has_builtin so the file compiles on any toolchain.
// ---------------------------------------------------------------------------

#ifndef __has_builtin
#define __has_builtin(x) 0
#endif

#if __has_builtin(__builtin_amdgcn_global_load_async_to_lds_b128)
#define HAVE_ASYNC 1
#else
#define HAVE_ASYNC 0
#endif

#if __has_builtin(__builtin_amdgcn_tensor_load_to_lds) && \
    __has_builtin(__builtin_amdgcn_s_wait_tensorcnt)
#define HAVE_TDM 1
#else
#define HAVE_TDM 0
#endif

typedef __attribute__((ext_vector_type(16))) __bf16 v16bf;
typedef __attribute__((ext_vector_type(8)))  float  v8f;

union FragB16 { v16bf v; uint4 u[2]; };

__device__ __forceinline__ float gelu_tanh(float x) {
    // jax.nn.gelu default (tanh approximation)
    float x3 = x * x * x;
    float t  = tanhf(0.7978845608028654f * (x + 0.044715f * x3));
    return 0.5f * x * (1.0f + t);
}

__device__ __forceinline__ float sigmoidf(float x) {
    return 1.0f / (1.0f + expf(-x));
}

#if HAVE_ASYNC
// 16-byte global -> LDS async copy (no VGPR round trip; tracked by ASYNCcnt).
// Builtin signature (from hipcc diagnostic): param0 = addrspace(1) int4*,
// param1 = addrspace(3) int4*, then imm offset + imm cpol.
typedef int i32x4a __attribute__((ext_vector_type(4)));
typedef __attribute__((address_space(1))) i32x4a* gp128_t;
typedef __attribute__((address_space(3))) i32x4a* lp128_t;
__device__ __forceinline__ void async_cp16(const void* g, void* l) {
    __builtin_amdgcn_global_load_async_to_lds_b128(
        (gp128_t)(uintptr_t)g, (lp128_t)l, 0, 0);
}
__device__ __forceinline__ void async_wait0() {
#if __has_builtin(__builtin_amdgcn_s_wait_asynccnt)
    __builtin_amdgcn_s_wait_asynccnt(0);
#else
    asm volatile("s_wait_asynccnt 0x0" ::: "memory");
#endif
}
#endif

#if HAVE_TDM
typedef unsigned int u32x4 __attribute__((ext_vector_type(4)));
typedef int          i32x8 __attribute__((ext_vector_type(8)));
typedef int          i32x4 __attribute__((ext_vector_type(4)));

// TDM: load a 128(row) x 32(col) bf16 tile from a row-major matrix with row
// stride K (elements) into LDS at lds_off, padding each 16-DWORD row chunk by
// 4 DWORDs (pad_interval=3 -> 64B, pad_amount=3 -> 16B) => 40 bf16 LDS stride.
// Descriptor packing per CDNA5 ISA 8.3/8.4 (count=1, type=2, data_size=2B).
__device__ __forceinline__ void tdm_load_tile_128x32(const __hip_bfloat16* gbase,
                                                     unsigned lds_off, int K) {
    unsigned long long ga = (unsigned long long)(uintptr_t)gbase;
    u32x4 g0;
    g0.x = 1u;                                         // count=1, no gather
    g0.y = lds_off;                                    // lds_addr [63:32]
    g0.z = (unsigned)(ga & 0xffffffffu);               // global_addr lo
    g0.w = (unsigned)((ga >> 32) & 0x01ffffffu) | (2u << 30);  // addr hi | type=2
    i32x8 g1;
    g1[0] = (1 << 16) | (1 << 20) | (3 << 22) | (3 << 25); // data_size=2B, pad_en,
                                                           // pad_interval=3, pad_amount=3
    g1[1] = (K & 0xffff) << 16;                        // tensor_dim0 lo16 @ [63:48]
    g1[2] = ((K >> 16) & 0xffff) | (128 << 16);        // tensor_dim0 hi | tensor_dim1 lo
    g1[3] = (32 << 16);                                // tensor_dim1 hi=0 | tile_dim0=32
    g1[4] = 128;                                       // tile_dim1=128, tile_dim2=0
    g1[5] = K;                                         // tensor_dim0_stride lo32
    g1[6] = 0;                                         // stride hi | dim1_stride lo
    g1[7] = 0;
    i32x4 z4 = {0, 0, 0, 0};
#if __clang_major__ >= 23
    i32x8 z8 = {0, 0, 0, 0, 0, 0, 0, 0};
    __builtin_amdgcn_tensor_load_to_lds(g0, g1, z4, z4, z8, 0);
#else
    __builtin_amdgcn_tensor_load_to_lds(g0, g1, z4, z4, 0);
#endif
}
#endif

// ---------------------------------------------------------------------------
// Weight convert + transpose: w[K,N] f32 -> wt[N,K] bf16
// ---------------------------------------------------------------------------
__global__ void cvt_transpose_kernel(const float* __restrict__ w,
                                     __hip_bfloat16* __restrict__ wt,
                                     int K, int N) {
    size_t gid = (size_t)blockIdx.x * blockDim.x + threadIdx.x;
    if (gid >= (size_t)K * N) return;
    int k = (int)(gid / N);
    int n = (int)(gid % N);
    wt[(size_t)n * K + k] = __float2bfloat16(w[gid]);
}

// ---------------------------------------------------------------------------
// RMSNorm: x[M,D] f32 -> out[M,D] bf16 (one block per row)
// ---------------------------------------------------------------------------
__global__ __launch_bounds__(256)
void rmsnorm_bf16_kernel(const float* __restrict__ x, const float* __restrict__ w,
                         __hip_bfloat16* __restrict__ out, int D) {
    __shared__ float red[256];
    int row = blockIdx.x;
    const float* xr = x + (size_t)row * D;
    float s = 0.0f;
    for (int c = threadIdx.x; c < D; c += 256) { float v = xr[c]; s += v * v; }
    red[threadIdx.x] = s;
    __syncthreads();
    for (int off = 128; off > 0; off >>= 1) {
        if (threadIdx.x < (unsigned)off) red[threadIdx.x] += red[threadIdx.x + off];
        __syncthreads();
    }
    float scale = rsqrtf(red[0] / (float)D + 1e-6f);
    for (int c = threadIdx.x; c < D; c += 256)
        out[(size_t)row * D + c] = __float2bfloat16(xr[c] * scale * w[c]);
}

// ---------------------------------------------------------------------------
// Causal depthwise conv, K=4. xb0[M,D] f32 -> xb f32 + xb_bf bf16
// ---------------------------------------------------------------------------
__global__ void conv_kernel(const float* __restrict__ xb0,
                            const float* __restrict__ cw,   // [4,D]
                            const float* __restrict__ cb,   // [D]
                            float* __restrict__ xb,
                            __hip_bfloat16* __restrict__ xb_bf,
                            int S, int D, size_t total) {
    size_t gid = (size_t)blockIdx.x * blockDim.x + threadIdx.x;
    if (gid >= total) return;
    int d = (int)(gid % D);
    int s = (int)((gid / D) % S);
    float acc = cb[d];
#pragma unroll
    for (int k = 0; k < 4; ++k) {
        int ss = s - 3 + k;
        if (ss >= 0) acc += cw[k * D + d] * xb0[gid + (size_t)((long)(ss - s) * D)];
    }
    xb[gid]    = acc;
    xb_bf[gid] = __float2bfloat16(acc);
}

// ---------------------------------------------------------------------------
// RG-LRU chunked scan (3 passes)
// ---------------------------------------------------------------------------
__device__ __forceinline__ void rglru_ab(float ral, float ial, float xv, float coef,
                                         float& a, float& b) {
    float r = sigmoidf(ral);
    float i = sigmoidf(ial);
    a = expf(coef * r);
    b = sqrtf(fmaxf(1.0f - a * a, 1e-12f)) * (i * xv);
}

__global__ void scan_pass1_kernel(const float* __restrict__ ra, const float* __restrict__ ia,
                                  const float* __restrict__ xb, const float* __restrict__ lam,
                                  float* __restrict__ Asum, float* __restrict__ Bsum,
                                  int Bb, int S, int D, int NC) {
    int CH = S / NC;
    size_t gid = (size_t)blockIdx.x * blockDim.x + threadIdx.x;
    size_t total = (size_t)Bb * NC * D;
    if (gid >= total) return;
    int d  = (int)(gid % D);
    int ch = (int)((gid / D) % NC);
    int b  = (int)(gid / ((size_t)D * NC));
    float l    = lam[d];
    float sp   = (l > 20.0f) ? l : log1pf(expf(l));
    float coef = -8.0f * sp;
    float aprod = 1.0f, bacc = 0.0f;
    size_t base = ((size_t)b * S + (size_t)ch * CH) * D + d;
    for (int s = 0; s < CH; ++s) {
        size_t idx = base + (size_t)s * D;
        float a, bv;
        rglru_ab(ra[idx], ia[idx], xb[idx], coef, a, bv);
        bacc  = a * bacc + bv;
        aprod = aprod * a;
    }
    Asum[gid] = aprod;
    Bsum[gid] = bacc;
}

__global__ void scan_pass2_kernel(const float* __restrict__ Asum, const float* __restrict__ Bsum,
                                  float* __restrict__ carry, int Bb, int D, int NC) {
    size_t gid = (size_t)blockIdx.x * blockDim.x + threadIdx.x;
    if (gid >= (size_t)Bb * D) return;
    int d = (int)(gid % D);
    int b = (int)(gid / D);
    float h = 0.0f;
    for (int ch = 0; ch < NC; ++ch) {
        size_t idx = ((size_t)b * NC + ch) * D + d;
        carry[idx] = h;
        h = Asum[idx] * h + Bsum[idx];
    }
}

__global__ void scan_pass3_kernel(const float* __restrict__ ra, const float* __restrict__ ia,
                                  const float* __restrict__ xb, const float* __restrict__ lam,
                                  const float* __restrict__ carry, float* __restrict__ h,
                                  int Bb, int S, int D, int NC) {
    int CH = S / NC;
    size_t gid = (size_t)blockIdx.x * blockDim.x + threadIdx.x;
    size_t total = (size_t)Bb * NC * D;
    if (gid >= total) return;
    int d  = (int)(gid % D);
    int ch = (int)((gid / D) % NC);
    int b  = (int)(gid / ((size_t)D * NC));
    float l    = lam[d];
    float sp   = (l > 20.0f) ? l : log1pf(expf(l));
    float coef = -8.0f * sp;
    float hprev = carry[gid];
    size_t base = ((size_t)b * S + (size_t)ch * CH) * D + d;
    for (int s = 0; s < CH; ++s) {
        size_t idx = base + (size_t)s * D;
        float a, bv;
        rglru_ab(ra[idx], ia[idx], xb[idx], coef, a, bv);
        hprev = a * hprev + bv;
        h[idx] = hprev;
    }
}

// ---------------------------------------------------------------------------
// Elementwise: out_bf = bf16(a * b)
// ---------------------------------------------------------------------------
__global__ void mul_bf16_kernel(const float* __restrict__ a, const float* __restrict__ b,
                                __hip_bfloat16* __restrict__ out, size_t n) {
    size_t gid = (size_t)blockIdx.x * blockDim.x + threadIdx.x;
    if (gid < n) out[gid] = __float2bfloat16(a[gid] * b[gid]);
}

// ---------------------------------------------------------------------------
// WMMA GEMM: C[M,N] = epi(A[M,K] @ Bt[N,K]^T)
// 256 threads = 8 waves; block tile 128x128; K-step 32; wave tile 64x32
// (4x2 sub-tiles of 16x16 -> 8 v8f accumulators).
// A tile staged via async global->LDS copies; Bt (weight) tile via TDM.
// ---------------------------------------------------------------------------
enum { EPI_F32 = 0, EPI_BF16 = 1, EPI_GELU_F32 = 2, EPI_GELU_BF16 = 3,
       EPI_MUL_BF16 = 4, EPI_RES_F32 = 5 };

#define LDS_STRIDE 40   // 32 bf16 + 8 pad (matches TDM pad_interval=3/pad_amount=3)

template <int EPI>
__global__ __launch_bounds__(256)
void wmma_gemm_kernel(const __hip_bfloat16* __restrict__ A,
                      const __hip_bfloat16* __restrict__ Bt,
                      void* __restrict__ Cout,
                      const float* __restrict__ res,
                      const __hip_bfloat16* __restrict__ aux,
                      int N, int K) {
    __shared__ unsigned short As[128][LDS_STRIDE];
    __shared__ unsigned short Bs[128][LDS_STRIDE];

    const int tid   = threadIdx.x;
    const int wave  = tid >> 5;
    const int lane  = tid & 31;
    const int waveM = wave >> 2;   // 0..1 -> 64 rows
    const int waveN = wave & 3;    // 0..3 -> 32 cols
    const int lr    = lane & 15;
    const int hsel  = lane >> 4;   // half-wave select

    const int m0 = blockIdx.y * 128;
    const int n0 = blockIdx.x * 128;

#if HAVE_TDM
    typedef __attribute__((address_space(3))) void lvoid3;
    const unsigned ldsB = (unsigned)(uintptr_t)(lvoid3*)&Bs[0][0];
#endif

    v8f acc[4][2];
#pragma unroll
    for (int i = 0; i < 4; ++i)
#pragma unroll
        for (int j = 0; j < 2; ++j) acc[i][j] = (v8f)(0.0f);

    for (int k0 = 0; k0 < K; k0 += 32) {
        // ---- stage A (and possibly B) tiles into LDS ----
#if HAVE_TDM
        if (wave == 0) {
            tdm_load_tile_128x32(Bt + (size_t)n0 * K + k0, ldsB, K);
        }
#endif
#if HAVE_ASYNC
#pragma unroll
        for (int i = 0; i < 2; ++i) {
            int c   = tid + 256 * i;
            int row = c >> 2;
            int seg = c & 3;
            async_cp16(A + (size_t)(m0 + row) * K + k0 + seg * 8, &As[row][seg * 8]);
#if !HAVE_TDM
            async_cp16(Bt + (size_t)(n0 + row) * K + k0 + seg * 8, &Bs[row][seg * 8]);
#endif
        }
#else
#pragma unroll
        for (int i = 0; i < 2; ++i) {
            int c   = tid + 256 * i;
            int row = c >> 2;
            int seg = c & 3;
            *(uint4*)&As[row][seg * 8] =
                *(const uint4*)(A + (size_t)(m0 + row) * K + k0 + seg * 8);
#if !HAVE_TDM
            *(uint4*)&Bs[row][seg * 8] =
                *(const uint4*)(Bt + (size_t)(n0 + row) * K + k0 + seg * 8);
#endif
        }
#endif
        if (k0 + 32 < K) {  // CDNA5 global_prefetch_b8 on next K tile
            int c   = tid;
            int row = c >> 2;
            int seg = c & 3;
            __builtin_prefetch(A  + (size_t)(m0 + row) * K + k0 + 32 + seg * 8, 0, 1);
            __builtin_prefetch(Bt + (size_t)(n0 + row) * K + k0 + 32 + seg * 8, 0, 1);
        }
#if HAVE_TDM
        if (wave == 0) __builtin_amdgcn_s_wait_tensorcnt(0);
#endif
#if HAVE_ASYNC
        async_wait0();
#endif
        __syncthreads();

        // ---- fragments + WMMA ----
        FragB16 bfrag[2];
#pragma unroll
        for (int ns = 0; ns < 2; ++ns) {
            int n  = waveN * 32 + ns * 16 + lr;
            int kb = hsel * 16;
            bfrag[ns].u[0] = *(const uint4*)&Bs[n][kb];
            bfrag[ns].u[1] = *(const uint4*)&Bs[n][kb + 8];
        }
#pragma unroll
        for (int ms = 0; ms < 4; ++ms) {
            int r  = waveM * 64 + ms * 16 + lr;
            int kb = hsel * 16;
            FragB16 afrag;
            afrag.u[0] = *(const uint4*)&As[r][kb];        // K 0..7  / 16..23
            afrag.u[1] = *(const uint4*)&As[r][kb + 8];    // K 8..15 / 24..31
#pragma unroll
            for (int ns = 0; ns < 2; ++ns) {
                acc[ms][ns] = __builtin_amdgcn_wmma_f32_16x16x32_bf16(
                    false, afrag.v, false, bfrag[ns].v,
                    (short)0, acc[ms][ns], false, false);
            }
        }
        __syncthreads();
    }

    // ---- epilogue ----
#pragma unroll
    for (int ms = 0; ms < 4; ++ms) {
#pragma unroll
        for (int ns = 0; ns < 2; ++ns) {
            int n = n0 + waveN * 32 + ns * 16 + lr;
#pragma unroll
            for (int j = 0; j < 8; ++j) {
                int m = m0 + waveM * 64 + ms * 16 + j + hsel * 8;
                size_t idx = (size_t)m * N + n;
                float c = acc[ms][ns][j];
                if constexpr (EPI == EPI_GELU_F32 || EPI == EPI_GELU_BF16) c = gelu_tanh(c);
                if constexpr (EPI == EPI_MUL_BF16) c *= __bfloat162float(aux[idx]);
                if constexpr (EPI == EPI_RES_F32)  c += res[idx];
                if constexpr (EPI == EPI_BF16 || EPI == EPI_GELU_BF16 || EPI == EPI_MUL_BF16)
                    ((__hip_bfloat16*)Cout)[idx] = __float2bfloat16(c);
                else
                    ((float*)Cout)[idx] = c;
            }
        }
    }
}

// ---------------------------------------------------------------------------
// Host orchestration
// ---------------------------------------------------------------------------
static inline size_t align256(size_t x) { return (x + 255) & ~(size_t)255; }

extern "C" void kernel_launch(void* const* d_in, const int* in_sizes, int n_in,
                              void* d_out, int out_size, void* d_ws, size_t ws_size,
                              hipStream_t stream) {
    const float* x       = (const float*)d_in[0];
    const float* norm1_w = (const float*)d_in[1];
    const float* Wx      = (const float*)d_in[2];
    const float* Wy      = (const float*)d_in[3];
    const float* conv_w  = (const float*)d_in[4];
    const float* conv_b  = (const float*)d_in[5];
    const float* Wi      = (const float*)d_in[6];
    const float* Wa      = (const float*)d_in[7];
    const float* lam     = (const float*)d_in[8];
    const float* Wo      = (const float*)d_in[9];
    const float* norm2_w = (const float*)d_in[10];
    const float* Wg      = (const float*)d_in[11];
    const float* Wu      = (const float*)d_in[12];
    const float* Wd      = (const float*)d_in[13];
    float* out = (float*)d_out;

    const int D = in_sizes[1];                       // 1024
    const size_t MD = (size_t)in_sizes[0];           // B*S*D
    const int M = (int)(MD / D);                     // 8192
    const int H = in_sizes[11] / D;                  // 4096
    const int S = (M % 2048 == 0) ? 2048 : M;        // reference shapes
    const int Bb = M / S;
    const int NC = (S % 16 == 0) ? 16 : 1;           // scan chunks

    char* ws = (char*)d_ws;
    size_t off = 0;
    auto alloc = [&](size_t bytes) -> void* {
        void* p = ws + off; off = align256(off + bytes); return p;
    };
    typedef __hip_bfloat16 bf;
    bf* WxT  = (bf*)alloc((size_t)D * D * 2);
    bf* WyT  = (bf*)alloc((size_t)D * D * 2);
    bf* WaT  = (bf*)alloc((size_t)D * D * 2);
    bf* WiT  = (bf*)alloc((size_t)D * D * 2);
    bf* WoT  = (bf*)alloc((size_t)D * D * 2);
    bf* WgT  = (bf*)alloc((size_t)D * H * 2);
    bf* WuT  = (bf*)alloc((size_t)D * H * 2);
    bf* WdT  = (bf*)alloc((size_t)H * D * 2);
    bf* t_bf   = (bf*)alloc(MD * 2);
    float* xb0 = (float*)alloc(MD * 4);
    float* xb  = (float*)alloc(MD * 4);
    bf* xb_bf  = (bf*)alloc(MD * 2);
    float* ra  = (float*)alloc(MD * 4);
    float* ia  = (float*)alloc(MD * 4);
    float* hbuf= (float*)alloc(MD * 4);
    float* yb  = (float*)alloc(MD * 4);
    bf* hy_bf  = (bf*)alloc(MD * 2);
    float* h1  = (float*)alloc(MD * 4);
    bf* u_bf   = (bf*)alloc(MD * 2);
    bf* g_bf   = (bf*)alloc((size_t)M * H * 2);
    bf* gv_bf  = (bf*)alloc((size_t)M * H * 2);
    float* Asum  = (float*)alloc((size_t)Bb * NC * D * 4);
    float* Bsum  = (float*)alloc((size_t)Bb * NC * D * 4);
    float* carry = (float*)alloc((size_t)Bb * NC * D * 4);

    const dim3 blk(256);
    auto eb = [&](size_t n) { return dim3((unsigned)((n + 255) / 256)); };

    // 1) weight convert+transpose
    cvt_transpose_kernel<<<eb((size_t)D * D), blk, 0, stream>>>(Wx, WxT, D, D);
    cvt_transpose_kernel<<<eb((size_t)D * D), blk, 0, stream>>>(Wy, WyT, D, D);
    cvt_transpose_kernel<<<eb((size_t)D * D), blk, 0, stream>>>(Wa, WaT, D, D);
    cvt_transpose_kernel<<<eb((size_t)D * D), blk, 0, stream>>>(Wi, WiT, D, D);
    cvt_transpose_kernel<<<eb((size_t)D * D), blk, 0, stream>>>(Wo, WoT, D, D);
    cvt_transpose_kernel<<<eb((size_t)D * H), blk, 0, stream>>>(Wg, WgT, D, H);
    cvt_transpose_kernel<<<eb((size_t)D * H), blk, 0, stream>>>(Wu, WuT, D, H);
    cvt_transpose_kernel<<<eb((size_t)H * D), blk, 0, stream>>>(Wd, WdT, H, D);

    // 2) t = rmsnorm(x)
    rmsnorm_bf16_kernel<<<dim3(M), blk, 0, stream>>>(x, norm1_w, t_bf, D);

    // 3) xb0 = t @ Wx
    wmma_gemm_kernel<EPI_F32><<<dim3(D / 128, M / 128), blk, 0, stream>>>(
        t_bf, WxT, xb0, nullptr, nullptr, D, D);

    // 4) causal depthwise conv
    conv_kernel<<<eb(MD), blk, 0, stream>>>(xb0, conv_w, conv_b, xb, xb_bf, S, D, MD);

    // 5) gate logits ra = xb@Wa, ia = xb@Wi
    wmma_gemm_kernel<EPI_F32><<<dim3(D / 128, M / 128), blk, 0, stream>>>(
        xb_bf, WaT, ra, nullptr, nullptr, D, D);
    wmma_gemm_kernel<EPI_F32><<<dim3(D / 128, M / 128), blk, 0, stream>>>(
        xb_bf, WiT, ia, nullptr, nullptr, D, D);

    // 6) RG-LRU chunked scan
    scan_pass1_kernel<<<eb((size_t)Bb * NC * D), blk, 0, stream>>>(
        ra, ia, xb, lam, Asum, Bsum, Bb, S, D, NC);
    scan_pass2_kernel<<<eb((size_t)Bb * D), blk, 0, stream>>>(
        Asum, Bsum, carry, Bb, D, NC);
    scan_pass3_kernel<<<eb((size_t)Bb * NC * D), blk, 0, stream>>>(
        ra, ia, xb, lam, carry, hbuf, Bb, S, D, NC);

    // 7) yb = gelu(t @ Wy)
    wmma_gemm_kernel<EPI_GELU_F32><<<dim3(D / 128, M / 128), blk, 0, stream>>>(
        t_bf, WyT, yb, nullptr, nullptr, D, D);

    // 8) hy = bf16(h * yb)
    mul_bf16_kernel<<<eb(MD), blk, 0, stream>>>(hbuf, yb, hy_bf, MD);

    // 9) h1 = x + hy @ Wo
    wmma_gemm_kernel<EPI_RES_F32><<<dim3(D / 128, M / 128), blk, 0, stream>>>(
        hy_bf, WoT, h1, x, nullptr, D, D);

    // 10) u = rmsnorm(h1)
    rmsnorm_bf16_kernel<<<dim3(M), blk, 0, stream>>>(h1, norm2_w, u_bf, D);

    // 11) g = gelu(u @ Wg)   [M,H]
    wmma_gemm_kernel<EPI_GELU_BF16><<<dim3(H / 128, M / 128), blk, 0, stream>>>(
        u_bf, WgT, g_bf, nullptr, nullptr, H, D);

    // 12) gv = g * (u @ Wu)  [M,H]
    wmma_gemm_kernel<EPI_MUL_BF16><<<dim3(H / 128, M / 128), blk, 0, stream>>>(
        u_bf, WuT, gv_bf, nullptr, g_bf, H, D);

    // 13) out = h1 + gv @ Wd
    wmma_gemm_kernel<EPI_RES_F32><<<dim3(D / 128, M / 128), blk, 0, stream>>>(
        gv_bf, WdT, out, h1, nullptr, D, H);
}
